// NeighborNN_79577154060336
// MI455X (gfx1250) — matrix operand order repaired and checked by reading, not compiled, not run on hardware
//
#include <hip/hip_runtime.h>
#include <hip/hip_bf16.h>
#include <math.h>

// ---------------------------------------------------------------------------
// Problem constants (from reference)
// ---------------------------------------------------------------------------
#define BQ   4096     // queries
#define ND   65536    // data points
#define DD   128      // feature dim
#define KNB  32       // n_neighbors (== wave32 width!)
#define HH   512      // hidden
#define INF_ (4256)   // (D+1)*K + D

typedef unsigned short u16;
typedef __attribute__((ext_vector_type(16))) __bf16 v16bf;
typedef __attribute__((ext_vector_type(8)))  float  v8f;

// ---------------------------------------------------------------------------
// bf16 helpers (RNE)
// ---------------------------------------------------------------------------
__device__ __forceinline__ u16 f32_to_bf16(float f) {
  unsigned int u = __float_as_uint(f);
  unsigned int r = u + 0x7FFFu + ((u >> 16) & 1u);
  return (u16)(r >> 16);
}
__device__ __forceinline__ float bf16_to_f32(u16 h) {
  return __uint_as_float(((unsigned int)h) << 16);
}

// ---------------------------------------------------------------------------
// WMMA fragments per CDNA5 ISA 7.12.2 (wave32)
// A 16x32 bf16: lane<16 -> row M=lane, K = {2v,2v+1} (v<4) / {16+2(v-4)..} ; +8 for lanes>=16
// B 32x16 bf16: lane<16 -> col N=lane, K = {2v,2v+1} (v<4) / {16+2(v-4)..} ; +8 for lanes>=16
// ---------------------------------------------------------------------------
struct Frag { union { v16bf v; unsigned int u[8]; }; };

__device__ __forceinline__ void load_a(Frag& f, const u16* A, int lda,
                                       int rowBase, int kBase, int lane) {
  const int half = lane >> 4, r = lane & 15;
  const u16* p = A + (size_t)(rowBase + r) * lda + kBase + 8 * half;
#pragma unroll
  for (int v = 0; v < 4; ++v) f.u[v]     = *(const unsigned int*)(p + 2 * v);
#pragma unroll
  for (int v = 0; v < 4; ++v) f.u[4 + v] = *(const unsigned int*)(p + 16 + 2 * v);
}

__device__ __forceinline__ void load_b(Frag& f, const u16* Bt, int ldb,
                                       int colBase, int kBase, int lane) {
  const int half = lane >> 4, c = lane & 15;
  const u16* p = Bt + (size_t)(colBase + c) * ldb + kBase + 8 * half;
#pragma unroll
  for (int v = 0; v < 4; ++v) f.u[v]     = *(const unsigned int*)(p + 2 * v);
#pragma unroll
  for (int v = 0; v < 4; ++v) f.u[4 + v] = *(const unsigned int*)(p + 16 + 2 * v);
}

__device__ __forceinline__ v8f wmma_bf16(const Frag& a, const Frag& b, v8f c) {
  return __builtin_amdgcn_wmma_f32_16x16x32_bf16(
      /*neg_a=*/false, a.v, /*neg_b=*/false, b.v,
      /*c_mod=*/(short)0, c, /*reuse_a=*/false, /*reuse_b=*/false);
}

// ---------------------------------------------------------------------------
// Async global -> LDS 16-byte copy (CDNA5 GLOBAL_LOAD_ASYNC_TO_LDS_B128,
// tracked with ASYNCcnt).  LDS byte address = low 32 bits of flat shared ptr
// (ISA 10.2).
// ---------------------------------------------------------------------------
__device__ __forceinline__ void async_copy_b128(void* lds_dst, const void* gsrc) {
  unsigned int lds_off = (unsigned int)(unsigned long long)lds_dst;
  unsigned long long ga = (unsigned long long)gsrc;
  asm volatile("global_load_async_to_lds_b128 %0, %1, off"
               :: "v"(lds_off), "v"(ga) : "memory");
}
__device__ __forceinline__ void wait_asynccnt0() {
  asm volatile("s_wait_asynccnt 0x0" ::: "memory");
}

// ---------------------------------------------------------------------------
// Kernel 1: f32 rows -> bf16 rows + squared norms. One wave per 128-elem row.
// Rows [0, BQ) come from x, rows [BQ, BQ+ND) from X_data.
// ---------------------------------------------------------------------------
__global__ __launch_bounds__(256) void convert_rows_kernel(
    const float* __restrict__ x, const float* __restrict__ X,
    u16* __restrict__ xb, u16* __restrict__ Xb,
    float* __restrict__ xn, float* __restrict__ Xn) {
  const int gw   = (blockIdx.x * blockDim.x + threadIdx.x) >> 5;  // global wave = row
  const int lane = threadIdx.x & 31;
  if (gw >= BQ + ND) return;
  const float* src; u16* dst; float* nrm; int row;
  if (gw < BQ) { row = gw;      src = x + (size_t)row * DD; dst = xb + (size_t)row * DD; nrm = xn + row; }
  else         { row = gw - BQ; src = X + (size_t)row * DD; dst = Xb + (size_t)row * DD; nrm = Xn + row; }

  float4 f = ((const float4*)src)[lane];                 // 4 consecutive floats
  unsigned int u0 = (unsigned)f32_to_bf16(f.x) | ((unsigned)f32_to_bf16(f.y) << 16);
  unsigned int u1 = (unsigned)f32_to_bf16(f.z) | ((unsigned)f32_to_bf16(f.w) << 16);
  ((uint2*)dst)[lane] = make_uint2(u0, u1);

  float s = f.x * f.x + f.y * f.y + f.z * f.z + f.w * f.w;
#pragma unroll
  for (int off = 16; off; off >>= 1) s += __shfl_xor(s, off, 32);
  if (lane == 0) *nrm = s;
}

// ---------------------------------------------------------------------------
// Kernel 2: transpose + convert weights: src f32 [Kd, Nn] -> dstT bf16 [Nn, Kd]
// ---------------------------------------------------------------------------
__global__ __launch_bounds__(256) void transpose_bf16_kernel(
    const float* __restrict__ src, u16* __restrict__ dstT, int Kd, int Nn) {
  __shared__ float tile[32][33];
  const int kb = blockIdx.x * 32, nb = blockIdx.y * 32;
  for (int i = threadIdx.y; i < 32; i += 8) {
    tile[i][threadIdx.x] = src[(size_t)(kb + i) * Nn + nb + threadIdx.x];
  }
  __syncthreads();
  for (int i = threadIdx.y; i < 32; i += 8) {
    dstT[(size_t)(nb + i) * Kd + kb + threadIdx.x] = f32_to_bf16(tile[threadIdx.x][i]);
  }
}

// ---------------------------------------------------------------------------
// Kernel 3: fused distance GEMM (WMMA bf16) + running top-K (K = 32 = wave32)
// Block: 512 threads = 16 waves; 16 queries per block; 16x256 candidate tile
// per iteration. Candidate list lives in registers: lane i holds slot i.
// ---------------------------------------------------------------------------
#define NTILE 256
__global__ __launch_bounds__(512) void dist_topk_kernel(
    const u16* __restrict__ xb, const u16* __restrict__ Xb,
    const float* __restrict__ xn, const float* __restrict__ Xn,
    int* __restrict__ idxOut) {
  __shared__ float s_d2[16][NTILE + 1];
  __shared__ float s_xn[16];

  const int lane  = threadIdx.x & 31;
  const int w     = threadIdx.x >> 5;       // wave id 0..15
  const int mBase = blockIdx.x * 16;
  const int half  = lane >> 4, cl = lane & 15;

  if (threadIdx.x < 16) s_xn[threadIdx.x] = xn[mBase + threadIdx.x];

  // Query A-fragments stay resident in registers for the whole kernel.
  Frag Afrag[4];
#pragma unroll
  for (int kc = 0; kc < 4; ++kc) load_a(Afrag[kc], xb, DD, mBase, kc * 32, lane);

  float regD = __builtin_inff();  int regI  = 0;   // lane-private candidate slot
  float worst = __builtin_inff(); int slot = 0;    // wave-uniform max of the K best

  __syncthreads();

  for (int t = 0; t < ND / NTILE; ++t) {
    const int nBase0 = t * NTILE;
    const int nBase  = nBase0 + w * 16;

    // ---- phase 1: each wave computes its 16x16 distance tile via WMMA ----
    v8f acc;
#pragma unroll
    for (int i = 0; i < 8; ++i) acc[i] = 0.0f;
    Frag Bfrag;
#pragma unroll
    for (int kc = 0; kc < 4; ++kc) {
      load_b(Bfrag, Xb, DD, nBase, kc * 32, lane);
      acc = wmma_bf16(Afrag[kc], Bfrag, acc);
    }
    const float xnc = Xn[nBase + cl];
#pragma unroll
    for (int v = 0; v < 8; ++v) {
      const int r = v + 8 * half;
      s_d2[r][w * 16 + cl] = s_xn[r] + xnc - 2.0f * acc[v];
    }
    __syncthreads();

    // ---- phase 2: wave w owns query row w; ballot-filtered insert ----
#pragma unroll 1
    for (int c0 = 0; c0 < NTILE / 32; ++c0) {
      const float d = s_d2[w][c0 * 32 + lane];
      unsigned int mask = (unsigned int)__ballot(d < worst);
      while (mask) {
        const int j = __builtin_ctz(mask);
        mask &= mask - 1;
        const float dj = __shfl(d, j, 32);
        if (dj < worst) {
          if (lane == slot) { regD = dj; regI = nBase0 + c0 * 32 + j; }
          // wave-wide argmax over the 32 register slots
          float mv = regD; int ms = lane;
#pragma unroll
          for (int off = 16; off; off >>= 1) {
            const float ov = __shfl_xor(mv, off, 32);
            const int   os = __shfl_xor(ms, off, 32);
            if (ov > mv || (ov == mv && os < ms)) { mv = ov; ms = os; }
          }
          worst = mv; slot = ms;
        }
      }
    }
    __syncthreads();
  }

  // ---- rank sort (ascending distance, stable by slot) and emit indices ----
  int rank = 0;
#pragma unroll
  for (int j = 0; j < 32; ++j) {
    const float dj = __shfl(regD, j, 32);
    if (dj < regD || (dj == regD && j < lane)) rank++;
  }
  idxOut[(size_t)(mBase + w) * KNB + rank] = regI;
}

// ---------------------------------------------------------------------------
// Kernel 4: gather neighbor features: o0[m] = [ xb[m] | (Xb[idx], y[idx]) x32 ]
// ---------------------------------------------------------------------------
__global__ __launch_bounds__(256) void build_features_kernel(
    const u16* __restrict__ xb, const u16* __restrict__ Xb,
    const float* __restrict__ y, const int* __restrict__ idx,
    u16* __restrict__ o0) {
  const int m = blockIdx.x;
  __shared__ int s_i[KNB];
  if (threadIdx.x < KNB) s_i[threadIdx.x] = idx[(size_t)m * KNB + threadIdx.x];
  __syncthreads();
  u16* dst = o0 + (size_t)m * INF_;
  for (int e = threadIdx.x; e < INF_; e += 256) {
    u16 v;
    if (e < DD) {
      v = xb[(size_t)m * DD + e];
    } else {
      const int q = e - DD;
      const int j = q / (DD + 1);
      const int p = q - j * (DD + 1);
      const int n = s_i[j];
      v = (p < DD) ? Xb[(size_t)n * DD + p] : f32_to_bf16(y[n]);
    }
    dst[e] = v;
  }
}

// ---------------------------------------------------------------------------
// Kernel 5/6: C = tanh(A[M,K] * Bt[N,K]^T + bias), bf16 in / bf16 out, f32 acc.
// Block = 512 threads = 16 waves in a 2(M) x 8(N) arrangement -> 32x128 tile.
// A-tile (32x32, 2KB) and B-tile (128x32, 8KB) per K-step are staged into LDS
// with double-buffered GLOBAL_LOAD_ASYNC_TO_LDS_B128 (ASYNCcnt), so the copy
// of step k+1 overlaps the WMMA of step k and waves share tiles through LDS.
// ---------------------------------------------------------------------------
__global__ __launch_bounds__(512) void gemm_bias_tanh_kernel(
    const u16* __restrict__ A, const u16* __restrict__ Bt,
    const float* __restrict__ bias, u16* __restrict__ Out,
    int Nn, int Kd, int lda, int ldb) {
  __shared__ u16 sA[2][32 * 32];    // [buf][m*32 + k]
  __shared__ u16 sB[2][128 * 32];   // [buf][n*32 + k]

  const int tid   = threadIdx.x;
  const int lane  = tid & 31;
  const int wv    = tid >> 5;
  const int mBlk  = blockIdx.y * 32;
  const int nBlk  = blockIdx.x * 128;
  const int mSub  = (wv >> 3) * 16;           // wave's 16-row slice in the A tile
  const int nSub  = (wv & 7) * 16;            // wave's 16-col slice in the B tile

  // per-thread async copy assignments (16B chunks; 32 bf16 per row = 4 chunks)
  const int bn = tid >> 2, bq = tid & 3;      // 512 chunks of the B tile
  const int am = tid >> 2, aq = tid & 3;      // 128 chunks of the A tile (tid<128)

  auto issue_copy = [&](int buf, int k0) {
    async_copy_b128(&sB[buf][bn * 32 + bq * 8],
                    Bt + (size_t)(nBlk + bn) * ldb + k0 + bq * 8);
    if (tid < 128) {
      async_copy_b128(&sA[buf][am * 32 + aq * 8],
                      A + (size_t)(mBlk + am) * lda + k0 + aq * 8);
    }
  };

  v8f acc;
#pragma unroll
  for (int i = 0; i < 8; ++i) acc[i] = 0.0f;

  issue_copy(0, 0);
  wait_asynccnt0();
  __syncthreads();

  Frag a, b;
  for (int k0 = 0; k0 < Kd; k0 += 32) {
    const int cur = (k0 >> 5) & 1;
    if (k0 + 32 < Kd) issue_copy(cur ^ 1, k0 + 32);   // overlap copy with WMMA

    load_a(a, &sA[cur][0], 32, mSub, 0, lane);        // ds_load from LDS
    load_b(b, &sB[cur][0], 32, nSub, 0, lane);
    acc = wmma_bf16(a, b, acc);

    wait_asynccnt0();                                 // next buffer resident
    __syncthreads();                                  // all waves done with cur
  }

  const int half = lane >> 4;
  const int col  = nBlk + nSub + (lane & 15);
  const float bv = bias[col];
#pragma unroll
  for (int v = 0; v < 8; ++v) {
    const int row = mBlk + mSub + v + 8 * half;
    Out[(size_t)row * Nn + col] = f32_to_bf16(tanhf(acc[v] + bv));
  }
}

// ---------------------------------------------------------------------------
// Kernel 7: out[m] = sigmoid(dot(h2[m], Wlast) + blast).  One wave per row.
// ---------------------------------------------------------------------------
__global__ __launch_bounds__(256) void final_kernel(
    const u16* __restrict__ h2, const float* __restrict__ Wl,
    const float* __restrict__ bl, float* __restrict__ out) {
  const int gw   = blockIdx.x * 8 + (threadIdx.x >> 5);
  const int lane = threadIdx.x & 31;
  if (gw >= BQ) return;
  const u16* h = h2 + (size_t)gw * HH;
  float s = 0.0f;
  for (int i = lane; i < HH; i += 32) s += bf16_to_f32(h[i]) * Wl[i];
#pragma unroll
  for (int off = 16; off; off >>= 1) s += __shfl_xor(s, off, 32);
  if (lane == 0) out[gw] = 1.0f / (1.0f + expf(-(s + bl[0])));
}

// ---------------------------------------------------------------------------
// Host launcher
// ---------------------------------------------------------------------------
extern "C" void kernel_launch(void* const* d_in, const int* in_sizes, int n_in,
                              void* d_out, int out_size, void* d_ws, size_t ws_size,
                              hipStream_t stream) {
  const float* x  = (const float*)d_in[0];   // [4096,128]
  const float* X  = (const float*)d_in[1];   // [65536,128]
  const float* y  = (const float*)d_in[2];   // [65536]
  const float* W1 = (const float*)d_in[3];   // [4256,512]
  const float* b1 = (const float*)d_in[4];   // [512]
  const float* Wh = (const float*)d_in[5];   // [512,512]
  const float* bh = (const float*)d_in[6];   // [512]
  const float* Wl = (const float*)d_in[7];   // [512,1]
  const float* bl = (const float*)d_in[8];   // [1]
  float* out = (float*)d_out;                // [4096]

  // workspace carve-up (256B aligned)
  char* base = (char*)d_ws;
  size_t off = 0;
  auto alloc = [&](size_t bytes) -> void* {
    void* p = base + off;
    off = (off + bytes + 255) & ~(size_t)255;
    return p;
  };
  u16*   xb   = (u16*)  alloc((size_t)BQ * DD * 2);
  u16*   Xb   = (u16*)  alloc((size_t)ND * DD * 2);
  float* xn   = (float*)alloc((size_t)BQ * 4);
  float* Xn   = (float*)alloc((size_t)ND * 4);
  u16*   W1t  = (u16*)  alloc((size_t)HH * INF_ * 2);
  u16*   Wht  = (u16*)  alloc((size_t)HH * HH * 2);
  int*   idxb = (int*)  alloc((size_t)BQ * KNB * 4);
  u16*   o0   = (u16*)  alloc((size_t)BQ * INF_ * 2);
  u16*   h1   = (u16*)  alloc((size_t)BQ * HH * 2);
  u16*   h2   = (u16*)  alloc((size_t)BQ * HH * 2);
  (void)ws_size; (void)in_sizes; (void)n_in; (void)out_size;

  // 1) convert inputs to bf16 + row norms (one wave per row)
  convert_rows_kernel<<<(BQ + ND) / 8, 256, 0, stream>>>(x, X, xb, Xb, xn, Xn);

  // 2) weight transposes (f32 -> bf16, [K,N] -> [N,K])
  transpose_bf16_kernel<<<dim3(INF_ / 32, HH / 32), dim3(32, 8), 0, stream>>>(W1, W1t, INF_, HH);
  transpose_bf16_kernel<<<dim3(HH / 32, HH / 32), dim3(32, 8), 0, stream>>>(Wh, Wht, HH, HH);

  // 3) fused WMMA distance GEMM + top-32
  dist_topk_kernel<<<BQ / 16, 512, 0, stream>>>(xb, Xb, xn, Xn, idxb);

  // 4) gather features
  build_features_kernel<<<BQ, 256, 0, stream>>>(xb, Xb, y, idxb, o0);

  // 5) layer 1: [4096,4256] x [4256,512] -> tanh  (async-LDS staged WMMA)
  gemm_bias_tanh_kernel<<<dim3(HH / 128, BQ / 32), 512, 0, stream>>>(
      o0, W1t, b1, h1, HH, INF_, INF_, INF_);

  // 6) layer 2: [4096,512] x [512,512] -> tanh
  gemm_bias_tanh_kernel<<<dim3(HH / 128, BQ / 32), 512, 0, stream>>>(
      h1, Wht, bh, h2, HH, HH, HH, HH);

  // 7) output head: dot + sigmoid
  final_kernel<<<BQ / 8, 256, 0, stream>>>(h2, Wl, bl, out);
}
